// Model_78271484002702
// MI455X (gfx1250) — matrix operand order repaired and checked by reading, not compiled, not run on hardware
//
#include <hip/hip_runtime.h>
#include <hip/hip_bf16.h>
#include <cstdint>

// ---------------------------------------------------------------------------
// Problem constants (match reference)
// ---------------------------------------------------------------------------
#define N_NODES   520000      // B * NPG   (multiple of 32: 16250 row-tiles)
#define N_GRAPHS  40
#define NPG_      13000
#define N_EDGES   900000
#define CPG_      900
#define M_CFG     36000       // B * CPG
#define NF_       140
#define FEAT_EMB_ 20
#define OP_EMB_   12
#define CFG_      18
#define IN_CH_    32
#define CONCAT_   50          // 20 + 12 + 18
#define MAXCH     256

typedef __attribute__((ext_vector_type(16))) __bf16       v16bf;
typedef __attribute__((ext_vector_type(8)))  float        v8f;
typedef __attribute__((ext_vector_type(4)))  unsigned int v4u;
typedef __attribute__((ext_vector_type(8)))  int          v8i;
typedef __attribute__((ext_vector_type(4)))  int          v4i;

#if defined(__has_builtin)
#if __has_builtin(__builtin_amdgcn_tensor_load_to_lds)
#define HAVE_TDM 1
#endif
#endif

// ---------------------------------------------------------------------------
// TDM 1-D copy: nelem f32 from contiguous global memory into LDS.
// D# layout per CDNA5 ISA:
//   group0: count=1 | lds_addr | global_addr[56:0] | type=2
//   group1: data_size=4B, tensor_dim0 = tile_dim0 = tensor_dim0_stride = nelem
//   groups 2/3: zero (<=2-D tensor)
// Wave-level op: EXEC ignored, tracked by TENSORcnt.
// ---------------------------------------------------------------------------
#ifdef HAVE_TDM
__device__ __forceinline__ void tdm_load_f32_1d(unsigned lds_byte_off,
                                                const float* gsrc,
                                                unsigned nelem) {
    unsigned long long ga = (unsigned long long)(uintptr_t)gsrc;
    v4u g0;
    g0[0] = 1u;                                            // count=1 (valid), user mode
    g0[1] = lds_byte_off;                                  // lds_addr (bytes)
    g0[2] = (unsigned)(ga & 0xffffffffu);                  // global_addr[31:0]
    g0[3] = (unsigned)((ga >> 32) & 0x1ffffffu)            // global_addr[56:32]
          | 0x80000000u;                                   // type=2 ("image")
    v8i g1;
    g1[0] = (int)(2u << 16);                               // data_size: 2 -> 4 bytes
    g1[1] = (int)((nelem & 0xffffu) << 16);                // tensor_dim0[15:0]
    g1[2] = (int)((nelem >> 16) & 0xffffu);                // tensor_dim0[31:16]
    g1[3] = (int)((nelem & 0xffffu) << 16);                // tile_dim0 (<=65535)
    g1[4] = 0;                                             // tile_dim1/2 unused
    g1[5] = (int)nelem;                                    // tensor_dim0_stride[31:0]
    g1[6] = 0;
    g1[7] = 0;
    v4i gz = {0, 0, 0, 0};
#if __clang_major__ >= 23
    v8i gz8 = {0, 0, 0, 0, 0, 0, 0, 0};
    __builtin_amdgcn_tensor_load_to_lds(g0, g1, gz, gz, gz8, 0);
#else
    __builtin_amdgcn_tensor_load_to_lds(g0, g1, gz, gz, 0);
#endif
}
#endif

// ---------------------------------------------------------------------------
// Generic fill
// ---------------------------------------------------------------------------
__global__ void fill_f32(float* p, float v, size_t n) {
    size_t i      = (size_t)blockIdx.x * blockDim.x + threadIdx.x;
    size_t stride = (size_t)gridDim.x * blockDim.x;
    for (; i < n; i += stride) p[i] = v;
}

// ---------------------------------------------------------------------------
// Scatter per-node config features (rest stays zero from fill)
// ---------------------------------------------------------------------------
__global__ void scatter_cfg(const float* __restrict__ ncf,
                            const int*   __restrict__ ids,
                            const int*   __restrict__ ptr,
                            float*       __restrict__ cfg) {
    int i = blockIdx.x * blockDim.x + threadIdx.x;
    if (i >= M_CFG) return;
    int g    = i / CPG_;                 // node_config_ptr is uniform CPG
    int node = ids[i] + ptr[g];
    const float* s = ncf + (size_t)i * CFG_;
    float*       d = cfg + (size_t)node * CFG_;
    #pragma unroll
    for (int j = 0; j < CFG_; ++j) d[j] = s[j];
}

// ---------------------------------------------------------------------------
// Fused: log1p(relu(x)) @ w_feat + opcode emb + cfg -> concat @ w_in -> relu
// ---------------------------------------------------------------------------
__global__ void __launch_bounds__(256) embed_kernel(
        const float* __restrict__ node_feat,
        const int*   __restrict__ node_opcode,
        const float* __restrict__ cfg,
        const float* __restrict__ w_feat, const float* __restrict__ b_feat,
        const float* __restrict__ opcode_emb,
        const float* __restrict__ w_in,   const float* __restrict__ b_in,
        float*       __restrict__ feat0) {
    __shared__ float s_wf[NF_ * FEAT_EMB_];
    __shared__ float s_wi[CONCAT_ * IN_CH_];
    __shared__ float s_bf[FEAT_EMB_];
    __shared__ float s_bi[IN_CH_];
    for (int i = threadIdx.x; i < NF_ * FEAT_EMB_; i += blockDim.x) s_wf[i] = w_feat[i];
    for (int i = threadIdx.x; i < CONCAT_ * IN_CH_; i += blockDim.x) s_wi[i] = w_in[i];
    if (threadIdx.x < FEAT_EMB_) s_bf[threadIdx.x] = b_feat[threadIdx.x];
    if (threadIdx.x < IN_CH_)    s_bi[threadIdx.x] = b_in[threadIdx.x];
    __syncthreads();

    int n = blockIdx.x * blockDim.x + threadIdx.x;
    if (n >= N_NODES) return;

    float acc[FEAT_EMB_];
    #pragma unroll
    for (int j = 0; j < FEAT_EMB_; ++j) acc[j] = s_bf[j];

    const float* nf = node_feat + (size_t)n * NF_;
    for (int f = 0; f < NF_; ++f) {
        float v = nf[f];
        v = v > 0.0f ? v : 0.0f;
        v = log1pf(v);
        #pragma unroll
        for (int j = 0; j < FEAT_EMB_; ++j) acc[j] += v * s_wf[f * FEAT_EMB_ + j];
    }

    float out[IN_CH_];
    #pragma unroll
    for (int c = 0; c < IN_CH_; ++c) out[c] = s_bi[c];
    #pragma unroll
    for (int j = 0; j < FEAT_EMB_; ++j) {
        float v = acc[j];
        #pragma unroll
        for (int c = 0; c < IN_CH_; ++c) out[c] += v * s_wi[j * IN_CH_ + c];
    }
    int op = node_opcode[n];
    const float* oe = opcode_emb + (size_t)op * OP_EMB_;
    #pragma unroll
    for (int j = 0; j < OP_EMB_; ++j) {
        float v = oe[j];
        #pragma unroll
        for (int c = 0; c < IN_CH_; ++c) out[c] += v * s_wi[(FEAT_EMB_ + j) * IN_CH_ + c];
    }
    const float* cf = cfg + (size_t)n * CFG_;
    #pragma unroll
    for (int j = 0; j < CFG_; ++j) {
        float v = cf[j];
        #pragma unroll
        for (int c = 0; c < IN_CH_; ++c) out[c] += v * s_wi[(FEAT_EMB_ + OP_EMB_ + j) * IN_CH_ + c];
    }
    float* o = feat0 + (size_t)n * IN_CH_;
    #pragma unroll
    for (int c = 0; c < IN_CH_; ++c) o[c] = fmaxf(out[c], 0.0f);
}

// ---------------------------------------------------------------------------
// Edge aggregation: agg[tgt] += feat[src]; deg[tgt] += 1  (f32 global atomics)
// ---------------------------------------------------------------------------
__global__ void edge_agg(const int* __restrict__ src, const int* __restrict__ tgt,
                         const float* __restrict__ feat,
                         float* __restrict__ agg, float* __restrict__ deg,
                         int in_ch, int chunks) {
    long id    = (long)blockIdx.x * blockDim.x + threadIdx.x;
    long total = (long)N_EDGES * chunks;
    if (id >= total) return;
    int e = (int)(id / chunks);
    int c = (int)(id - (long)e * chunks);
    int s = src[e], t = tgt[e];
    const float* fs = feat + (size_t)s * in_ch + c * 8;
    float*       at = agg  + (size_t)t * in_ch + c * 8;
    #pragma unroll
    for (int j = 0; j < 8; ++j) atomicAdd(&at[j], fs[j]);
    if (c == 0) atomicAdd(&deg[t], 1.0f);
}

// ---------------------------------------------------------------------------
// Pre-pack Wcat = [w_l ; w_r] ((2*in_ch) x out_ch, f32) into fragment-major
// bf16 so the GEMM's B fragment is one contiguous 32-byte load per lane:
//   pb[(g*out_ch + n)*16 + j] = bf16( Wcat[g*16 + j][n] ),  g = kt*2 + hi
// ---------------------------------------------------------------------------
__global__ void pack_weights(const float* __restrict__ w_l,
                             const float* __restrict__ w_r,
                             __bf16* __restrict__ pb, int in_ch, int out_ch) {
    int idx   = blockIdx.x * blockDim.x + threadIdx.x;
    int total = 2 * in_ch * out_ch;
    if (idx >= total) return;
    int j = idx & 15;
    int t = idx >> 4;            // g*out_ch + n
    int n = t % out_ch;
    int g = t / out_ch;
    int k = g * 16 + j;          // row of Wcat
    float v = (k < in_ch) ? w_l[(size_t)k * out_ch + n]
                          : w_r[(size_t)(k - in_ch) * out_ch + n];
    pb[idx] = (__bf16)v;
}

// ---------------------------------------------------------------------------
// SAGE conv GEMM via WMMA bf16:
//   out = relu( [agg/max(deg,1) | feat] @ [w_l ; w_r] + b_l )
// Block = 256 threads (8 waves), owns 32 contiguous rows (two 16-row M tiles).
// Tile ingress: wave 0 issues two 1-D TDM DMAs (agg rows, feat rows -> LDS
// f32 staging), s_wait_tensorcnt 0, barrier; all waves convert to the bf16
// WMMA A-tile. Wave w owns 16 columns; one B fragment (32B contiguous load
// from fragment-major packed weights) feeds 2 wmma per K-step.
// ---------------------------------------------------------------------------
__global__ void __launch_bounds__(256) sage_gemm_wmma(
        const float*  __restrict__ feat, const float* __restrict__ agg,
        const float*  __restrict__ deg,
        const __bf16* __restrict__ pb,   const float* __restrict__ b_l,
        float* __restrict__ out, int in_ch, int ic_log2, int out_ch) {
    const int K2 = 2 * in_ch;                      // 64 .. 512, multiple of 32
    __shared__ __bf16 A_lds[32][2 * MAXCH + 8];    // row stride 1040B (16B mult)
    __shared__ float  stage[2][32 * MAXCH];        // TDM f32 landing zone
    __shared__ float  rdeg[32];

    const int row0 = blockIdx.x * 32;              // N_NODES % 32 == 0: no OOB rows
    const int tid  = threadIdx.x;
    const int wave = tid >> 5;
    const int lane = tid & 31;
    const unsigned tile_elems = (unsigned)(32 << ic_log2);   // 32 * in_ch

    if (tid < 32) rdeg[tid] = 1.0f / fmaxf(deg[row0 + tid], 1.0f);

#ifdef HAVE_TDM
    if (wave == 0) {                               // wave-level DMA, EXEC ignored
        // Truncated flat address of __shared__ == LDS byte offset (ISA 10.2)
        tdm_load_f32_1d((unsigned)(uintptr_t)&stage[0][0],
                        agg + (size_t)row0 * in_ch, tile_elems);
        tdm_load_f32_1d((unsigned)(uintptr_t)&stage[1][0],
                        feat + (size_t)row0 * in_ch, tile_elems);
        __builtin_amdgcn_s_wait_tensorcnt(0);
    }
    __syncthreads();
    // Convert staged f32 -> bf16 A-tile ([mean | feat] per row)
    for (unsigned idx = tid; idx < tile_elems; idx += 256) {
        unsigned r = idx >> ic_log2;
        unsigned k = idx & (unsigned)(in_ch - 1);
        A_lds[r][k]         = (__bf16)(stage[0][idx] * rdeg[r]);
        A_lds[r][in_ch + k] = (__bf16)stage[1][idx];
    }
#else
    __syncthreads();
    for (unsigned idx = tid; idx < tile_elems; idx += 256) {
        unsigned r = idx >> ic_log2;
        unsigned k = idx & (unsigned)(in_ch - 1);
        A_lds[r][k]         = (__bf16)(agg [(size_t)row0 * in_ch + idx] * rdeg[r]);
        A_lds[r][in_ch + k] = (__bf16)feat[(size_t)row0 * in_ch + idx];
    }
#endif
    __syncthreads();

    const int col0 = blockIdx.y * 128 + wave * 16;
    if (col0 >= out_ch) return;                    // wave-uniform: EXEC all-1s

    const int m  = lane & 15;
    const int hi = lane >> 4;
    const int nc = col0 + m;
    const int ntiles = K2 >> 5;

    // Strength-reduced hot-loop pointers
    const __bf16* la0 = &A_lds[m][hi * 16];        // +16B-aligned runs, +64B/kt
    const __bf16* la1 = &A_lds[m + 16][hi * 16];
    const __bf16* pbl = pb + ((size_t)hi * out_ch + nc) * 16;
    const size_t  pstride = (size_t)2 * out_ch * 16;

    union AV { uint4 u4[2]; v16bf v; };
    v8f acc0 = {}, acc1 = {};

    for (int kt = 0; kt < ntiles; ++kt) {
        AV a0, a1;
        a0.u4[0] = *(const uint4*)(la0);
        a0.u4[1] = *(const uint4*)(la0 + 8);
        a1.u4[0] = *(const uint4*)(la1);
        a1.u4[1] = *(const uint4*)(la1 + 8);
        v16bf b  = *(const v16bf*)(pbl);
        // Branchless prefetch of next weight panel (global_prefetch_b8)
        const __bf16* pnx = (kt + 1 < ntiles) ? (pbl + pstride) : pbl;
        __builtin_prefetch(pnx, 0, 3);
        acc0 = __builtin_amdgcn_wmma_f32_16x16x32_bf16(
                   false, a0.v, false, b, (short)0, acc0, false, false);
        acc1 = __builtin_amdgcn_wmma_f32_16x16x32_bf16(
                   false, a1.v, false, b, (short)0, acc1, false, false);
        la0 += 32; la1 += 32; pbl += pstride;
    }

    float bias = b_l[nc];
    float* o0 = out + (size_t)(row0 + hi * 8)      * out_ch + nc;
    float* o1 = out + (size_t)(row0 + 16 + hi * 8) * out_ch + nc;
    #pragma unroll
    for (int r = 0; r < 8; ++r) {                  // C/D layout: VGPR r -> M=r / r+8
        o0[(size_t)r * out_ch] = fmaxf(acc0[r] + bias, 0.0f);
        o1[(size_t)r * out_ch] = fmaxf(acc1[r] + bias, 0.0f);
    }
}

// ---------------------------------------------------------------------------
// Head: per_node = feat(256) . w_out + b_out ; per-graph atomic reduction
// ---------------------------------------------------------------------------
__global__ void head_kernel(const float* __restrict__ feat,
                            const float* __restrict__ w_out,
                            const float* __restrict__ b_out,
                            float* __restrict__ pg) {
    int n = blockIdx.x * blockDim.x + threadIdx.x;
    if (n >= N_NODES) return;
    const float* f = feat + (size_t)n * MAXCH;
    float s = b_out[0];
    #pragma unroll 8
    for (int k = 0; k < MAXCH; ++k) s += f[k] * w_out[k];
    atomicAdd(&pg[n / NPG_], s);
}

// ---------------------------------------------------------------------------
// Finalize: scale to seconds, emit per_graph + triu pairwise diffs
// ---------------------------------------------------------------------------
__global__ void finalize_kernel(const float* __restrict__ pg, float* __restrict__ out) {
    __shared__ float lat[N_GRAPHS];
    int t = threadIdx.x;
    if (t < N_GRAPHS) lat[t] = 0.001f * pg[t];
    __syncthreads();
    if (t < N_GRAPHS) out[t] = lat[t];
    if (t < 140) {                                 // 5 groups * 28 pairs
        int g = t / 28, p = t % 28;
        int i = 0, rem = p, cnt = 7;               // triu_indices(8, k=1) row-major
        while (rem >= cnt) { rem -= cnt; ++i; --cnt; }
        int j = i + 1 + rem;
        out[N_GRAPHS + t] = lat[g * 8 + i] - lat[g * 8 + j];
    }
}

// ---------------------------------------------------------------------------
// Launch
// ---------------------------------------------------------------------------
extern "C" void kernel_launch(void* const* d_in, const int* in_sizes, int n_in,
                              void* d_out, int out_size, void* d_ws, size_t ws_size,
                              hipStream_t stream) {
    (void)in_sizes; (void)n_in; (void)out_size; (void)ws_size;

    const float* node_feat        = (const float*)d_in[0];
    const float* node_config_feat = (const float*)d_in[1];
    const float* w_feat           = (const float*)d_in[2];
    const float* b_feat           = (const float*)d_in[3];
    const float* opcode_emb       = (const float*)d_in[4];
    const float* w_in             = (const float*)d_in[5];
    const float* b_in             = (const float*)d_in[6];
    const float* wl[6]; const float* bl[6]; const float* wr[6];
    for (int i = 0; i < 6; ++i) {
        wl[i] = (const float*)d_in[7 + 3 * i];
        bl[i] = (const float*)d_in[8 + 3 * i];
        wr[i] = (const float*)d_in[9 + 3 * i];
    }
    const float* w_out   = (const float*)d_in[25];
    const float* b_out   = (const float*)d_in[26];
    const int* node_opcode = (const int*)d_in[27];
    const int* ptr   = (const int*)d_in[29];
    const int* ncids = (const int*)d_in[30];
    const int* edge  = (const int*)d_in[32];
    const int* src = edge;
    const int* tgt = edge + N_EDGES;

    // Workspace carve-out (256B aligned)
    char* ws = (char*)d_ws;
    size_t off = 0;
    auto carve = [&](size_t bytes) -> void* {
        void* p = ws + off;
        off += (bytes + 255) & ~(size_t)255;
        return p;
    };
    float*  feat_a = (float*)carve((size_t)N_NODES * MAXCH * 4);
    float*  feat_b = (float*)carve((size_t)N_NODES * MAXCH * 4);
    float*  aggb   = (float*)carve((size_t)N_NODES * MAXCH * 4);
    float*  degb   = (float*)carve((size_t)N_NODES * 4);
    float*  cfgb   = (float*)carve((size_t)N_NODES * CFG_ * 4);
    __bf16* pbb    = (__bf16*)carve((size_t)2 * MAXCH * MAXCH * 2); // packed weights
    float*  pgb    = (float*)carve(64 * 4);

    dim3 blk(256);

    // Config feature scatter (zero background)
    fill_f32<<<2048, blk, 0, stream>>>(cfgb, 0.0f, (size_t)N_NODES * CFG_);
    scatter_cfg<<<(M_CFG + 255) / 256, blk, 0, stream>>>(node_config_feat, ncids, ptr, cfgb);

    // Fused embedding + input linear
    embed_kernel<<<(N_NODES + 255) / 256, blk, 0, stream>>>(
        node_feat, node_opcode, cfgb, w_feat, b_feat, opcode_emb, w_in, b_in, feat_a);

    const int in_chs[6]  = {32, 64, 64, 128, 128, 256};
    const int ic_l2s[6]  = { 5,  6,  6,   7,   7,   8};
    const int out_chs[6] = {64, 64, 128, 128, 256, 256};
    float* cur = feat_a;
    float* nxt = feat_b;
    for (int l = 0; l < 6; ++l) {
        int ic = in_chs[l], oc = out_chs[l];
        fill_f32<<<4096, blk, 0, stream>>>(aggb, 0.0f, (size_t)N_NODES * ic);
        fill_f32<<<512,  blk, 0, stream>>>(degb, 0.0f, (size_t)N_NODES);
        int  chunks = ic / 8;
        long total  = (long)N_EDGES * chunks;
        edge_agg<<<(unsigned)((total + 255) / 256), blk, 0, stream>>>(
            src, tgt, cur, aggb, degb, ic, chunks);
        pack_weights<<<(2 * ic * oc + 255) / 256, blk, 0, stream>>>(wl[l], wr[l], pbb, ic, oc);
        dim3 grid((N_NODES + 31) / 32, (oc + 127) / 128);
        sage_gemm_wmma<<<grid, blk, 0, stream>>>(cur, aggb, degb, pbb, bl[l], nxt,
                                                 ic, ic_l2s[l], oc);
        float* t = cur; cur = nxt; nxt = t;
    }

    fill_f32<<<1, 64, 0, stream>>>(pgb, 0.0f, 64);
    head_kernel<<<(N_NODES + 255) / 256, blk, 0, stream>>>(cur, w_out, b_out, pgb);
    finalize_kernel<<<1, 192, 0, stream>>>(pgb, (float*)d_out);
}
